// LikelihoodModel_59657095741875
// MI455X (gfx1250) — compile-verified, matrix operand order checked by reading, not run on hardware
//
#include <hip/hip_runtime.h>
#include <hip/hip_bf16.h>
#include <stdint.h>

typedef _Float16 v8h  __attribute__((ext_vector_type(8)));
typedef _Float16 v16h __attribute__((ext_vector_type(16)));
typedef float    v8f  __attribute__((ext_vector_type(8)));

#define DT_F 0.01f
#define TT   200
#define RR   32
#define CC   64
#define KK   128
#define LL   10
#define TRTOT 6400   // TT*RR

// workspace byte offsets
#define OFF_FACT   0u        // factors: 2000 f32
#define OFF_SMALL  8192u     // 128 f32: [0]peak1 [16]peak2 [32]spA [48]spT [64]logSpA [80]logSpT [96]lgammaSpA [112]logpi
#define OFF_EXPSUM 9216u     // 64*16 f32
#define OFF_SUMY   16384u    // K*C f32
#define OFF_E      65536u    // K*LL*C f32
#define OFF_WFB    524288u   // C*16*TRTOT f16 (~13.1 MB)

__device__ __forceinline__ float softplus_f(float x) {
    return x > 20.f ? x : log1pf(expf(x));
}

__device__ __forceinline__ float digamma_f(float x) {
    float r = 0.f;
    while (x < 6.f) { r -= 1.f / x; x += 1.f; }
    float inv = 1.f / x, inv2 = inv * inv;
    return r + logf(x) - 0.5f * inv
           - inv2 * (1.f/12.f - inv2 * (1.f/120.f - inv2 * (1.f/252.f)));
}

// ---------------- Kernel 1: scalars / factors / peaks ----------------
__global__ void prep_kernel(const float* __restrict__ beta, const float* __restrict__ alpha,
                            const float* __restrict__ theta, const float* __restrict__ pi,
                            float* __restrict__ ws) {
    float* factors = ws;
    float* smallb  = ws + OFF_SMALL / 4;
    int tid = threadIdx.x;
    for (int i = tid; i < LL * TT; i += blockDim.x)
        factors[i] = softplus_f(beta[i]);
    __syncthreads();
    if (tid < LL) {
        const float* row = factors + tid * TT;
        int bi = 20; float bv = row[20];
        for (int i = 21; i < 70; ++i) if (row[i] > bv) { bv = row[i]; bi = i; }
        smallb[tid] = (float)bi * DT_F;
        bi = 120; bv = row[120];
        for (int i = 121; i < 170; ++i) if (row[i] > bv) { bv = row[i]; bi = i; }
        smallb[16 + tid] = (float)bi * DT_F;
        float sa = softplus_f(alpha[tid]);
        float st = softplus_f(theta[tid]);
        smallb[32 + tid] = sa;
        smallb[48 + tid] = st;
        smallb[64 + tid] = logf(sa);
        smallb[80 + tid] = logf(st);
        smallb[96 + tid] = lgammaf(sa);
    }
    if (tid == 0) {
        float z[LL]; z[0] = 0.f;
        for (int i = 1; i < LL; ++i) z[i] = pi[i - 1];
        float m = z[0];
        for (int i = 1; i < LL; ++i) m = fmaxf(m, z[i]);
        float s = 0.f;
        for (int i = 0; i < LL; ++i) s += expf(z[i] - m);
        float lse = m + logf(s);
        for (int i = 0; i < LL; ++i) smallb[112 + i] = z[i] - lse;
    }
}

// ---------------- Kernel 2: wf -> wfB (f16 B-operand layout) + exp-sums ----------------
__global__ void warp_kernel(const float* __restrict__ trial_off, const float* __restrict__ cfg_off,
                            float* __restrict__ ws) {
    const int c = blockIdx.x;   // 0..63
    const int l = blockIdx.y;   // 0..15 (rows 10..15 are zero padding)
    _Float16* wfB = (_Float16*)((char*)ws + OFF_WFB) + (size_t)(c * 16 + l) * TRTOT;
    const float* factors = ws;
    float* smallb = ws + OFF_SMALL / 4;
    float* expsum = ws + OFF_EXPSUM / 4;
    const int tid = threadIdx.x;
    __shared__ float red[256];

    if (l >= LL) {
        for (int i = tid; i < TRTOT; i += blockDim.x) wfB[i] = (_Float16)0.f;
        if (tid == 0) expsum[c * 16 + l] = 0.f;
        return;
    }
    const float p1 = smallb[l], p2 = smallb[16 + l];
    float eacc = 0.f;
    for (int i = tid; i < TRTOT; i += blockDim.x) {
        int t = i >> 5, r = i & 31;
        float wfv;
        bool warped = (t >= 20 && t < 70) || (t >= 120 && t < 170);
        if (!warped) {
            wfv = factors[l * TT + t];
        } else {
            int   seg   = (t >= 120) ? 1 : 0;
            int   p     = t - (seg ? 120 : 20);
            float left  = (float)(seg ? 120 : 20) * DT_F;
            float right = (float)(seg ? 170 : 70) * DT_F;
            float peak  = seg ? p2 : p1;
            int   j     = seg * LL + l;
            float off   = trial_off[(r * CC + c) * (2 * LL) + j] + cfg_off[c * (2 * LL) + j];
            float s = peak + off;
            if (s <= left)  s = left + DT_F;
            if (s >= right) s = right - DT_F;
            float lsp    = s - left;
            float lslope = (peak - left) / lsp;
            float rslope = (peak - right) / (s - right);
            float lst    = (float)p * DT_F;
            float wt     = (lst < lsp) ? (lst * lslope + left) : ((lst - lsp) * rslope + peak);
            float wi     = wt / DT_F;
            float fif = floorf(wi), cif = ceilf(wi);
            float cw = wi - fif;
            int fi = (int)fif, ci = (int)cif;
            fi = fi < 0 ? 0 : (fi > TT - 1 ? TT - 1 : fi);
            ci = ci < 0 ? 0 : (ci > TT - 1 ? TT - 1 : ci);
            wfv = factors[l * TT + fi] * (1.f - cw) + factors[l * TT + ci] * cw;
        }
        eacc += expf(wfv);
        wfB[i] = (_Float16)wfv;
    }
    red[tid] = eacc;
    __syncthreads();
    for (int s2 = 128; s2 > 0; s2 >>= 1) {
        if (tid < s2) red[tid] += red[tid + s2];
        __syncthreads();
    }
    if (tid == 0) expsum[c * 16 + l] = red[0];
}

// ---------------- Kernel 3: fused WMMA GEMM (sumYN) + sum_Y over one Y pass ----------------
__global__ void __launch_bounds__(256) gemm_kernel(const float* __restrict__ Y,
                                                   float* __restrict__ ws) {
    __shared__ uint32_t ytile[64 * 16 * 16];   // 64KB: [c][prow(16)][dw(16)] of packed 2xf16
    const _Float16* wfB = (const _Float16*)((const char*)ws + OFF_WFB);
    float* E    = ws + OFF_E / 4;
    float* sumY = ws + OFF_SUMY / 4;

    const int k0   = blockIdx.x * 16;
    const int tid  = threadIdx.x;
    const int wave = tid >> 5, lane = tid & 31;
    const int m    = lane & 15, hi = lane >> 4;   // fragment row / K-half
    const int c4   = tid & 15,  kL = tid >> 4;    // loader mapping

    v8f acc[8];
    #pragma unroll
    for (int i = 0; i < 8; ++i) acc[i] = (v8f){0.f, 0.f, 0.f, 0.f, 0.f, 0.f, 0.f, 0.f};
    float sy0 = 0.f, sy1 = 0.f, sy2 = 0.f, sy3 = 0.f;

    const float* yrow = Y + (size_t)(k0 + kL) * TRTOT * CC + c4 * 4;

    for (int g = 0; g < 4; ++g) {
        const int trBase = (blockIdx.y * 4 + g) * 32;
        // prefetch target: next chunk's rows (clamped to stay inside Y)
        int ntr = trBase + 32;
        if (ntr + 31 >= TRTOT) ntr = trBase;

        // --- load 16k x 32tr x 64c f32, convert to f16, stage in LDS ---
        for (int tp = 0; tp < 16; ++tp) {
            __builtin_prefetch(yrow + (size_t)(ntr + 2 * tp) * CC, 0, 1);
            const float4 v0 = *(const float4*)(yrow + (size_t)(trBase + 2 * tp) * CC);
            const float4 v1 = *(const float4*)(yrow + (size_t)(trBase + 2 * tp + 1) * CC);
            sy0 += v0.x + v1.x; sy1 += v0.y + v1.y;
            sy2 += v0.z + v1.z; sy3 += v0.w + v1.w;
            const float a0[4] = {v0.x, v0.y, v0.z, v0.w};
            const float a1[4] = {v1.x, v1.y, v1.z, v1.w};
            #pragma unroll
            for (int e = 0; e < 4; ++e) {
                const int c = c4 * 4 + e;
                const int prow = kL ^ (c & 15);   // swizzle k-row by c -> fewer bank conflicts
                union { _Float16 h[2]; uint32_t u; } pk;
                pk.h[0] = (_Float16)a0[e];
                pk.h[1] = (_Float16)a1[e];
                ytile[c * 256 + prow * 16 + tp] = pk.u;
            }
        }
        __syncthreads();

        // --- preload ALL fragments into distinct registers, then 8 back-to-back WMMAs ---
        const _Float16* yt = (const _Float16*)ytile;
        v16h afr[8], bfr[8];
        #pragma unroll
        for (int cc = 0; cc < 8; ++cc) {          // global B-fragments: one load clause
            const int c = wave * 8 + cc;
            const _Float16* brow = wfB + (size_t)(c * 16 + m) * TRTOT + trBase + hi * 8;
            union { v16h v; v8h h[2]; } ub;
            ub.h[0] = *(const v8h*)brow;
            ub.h[1] = *(const v8h*)(brow + 16);
            bfr[cc] = ub.v;
        }
        #pragma unroll
        for (int cc = 0; cc < 8; ++cc) {          // LDS A-fragments
            const int c = wave * 8 + cc;
            const int prow  = m ^ (c & 15);
            const int abase = c * 512 + prow * 32 + hi * 8;   // f16 index
            union { v16h v; v8h h[2]; } ua;
            ua.h[0] = *(const v8h*)&yt[abase];
            ua.h[1] = *(const v8h*)&yt[abase + 16];
            afr[cc] = ua.v;
        }
        #pragma unroll
        for (int cc = 0; cc < 8; ++cc) {
            acc[cc] = __builtin_amdgcn_wmma_f32_16x16x32_f16(
                          false, afr[cc], false, bfr[cc], (short)0, acc[cc], false, false);
        }
        __syncthreads();
    }

    // --- epilogue: D[m,n]: n = lane&15 (= l), m = j + 8*hi ---
    if (m < LL) {
        #pragma unroll
        for (int cc = 0; cc < 8; ++cc) {
            const int c = wave * 8 + cc;
            #pragma unroll
            for (int j = 0; j < 8; ++j) {
                const int mk = j + hi * 8;
                __hip_atomic_fetch_add(&E[((size_t)(k0 + mk) * LL + m) * CC + c], acc[cc][j],
                                       __ATOMIC_RELAXED, __HIP_MEMORY_SCOPE_AGENT);
            }
        }
    }
    float* syp = &sumY[(size_t)(k0 + kL) * CC + c4 * 4];
    __hip_atomic_fetch_add(syp + 0, sy0, __ATOMIC_RELAXED, __HIP_MEMORY_SCOPE_AGENT);
    __hip_atomic_fetch_add(syp + 1, sy1, __ATOMIC_RELAXED, __HIP_MEMORY_SCOPE_AGENT);
    __hip_atomic_fetch_add(syp + 2, sy2, __ATOMIC_RELAXED, __HIP_MEMORY_SCOPE_AGENT);
    __hip_atomic_fetch_add(syp + 3, sy3, __ATOMIC_RELAXED, __HIP_MEMORY_SCOPE_AGENT);
}

// ---------------- Kernel 4: U / softmax / W*term -> scalar ----------------
__global__ void finalize_kernel(const float* __restrict__ access, float* __restrict__ ws,
                                float* __restrict__ out) {
    const int idx = blockIdx.x * blockDim.x + threadIdx.x;  // 0..8191
    const int k = idx >> 6, c = idx & 63;
    const float* smallb = ws + OFF_SMALL / 4;
    const float* expsum = ws + OFF_EXPSUM / 4;
    const float* E      = ws + OFF_E / 4;
    const float* sumYp  = ws + OFF_SUMY / 4;

    const float sy = sumYp[k * CC + c];
    float U[LL], term[LL], acl[LL];
    for (int l = 0; l < LL; ++l) {
        const float ac   = access[((size_t)c * KK + k) * LL + l];
        const float spA  = smallb[32 + l], spT  = smallb[48 + l];
        const float lSpA = smallb[64 + l], lSpT = smallb[80 + l];
        const float gln  = smallb[96 + l], lpi  = smallb[112 + l];
        const float lt2    = DT_F * expsum[c * 16 + l] + spT;
        const float loglt2 = logf(lt2);
        const float lt1    = sy + spA;
        const float syn    = E[((size_t)k * LL + l) * CC + c] * ac;
        const float x = syn - lt1 * ac * loglt2 + spA * lSpT + sy * ac * lSpA + lpi;
        U[l] = expf(x);
        const float A = lt1 * ac / lt2;
        const float B = digamma_f(lt1) - loglt2;
        term[l] = syn - A * lt2 - gln + spA * (lSpT + B) + lpi;
        acl[l]  = ac;
    }
    float mu = U[0];
    for (int l = 1; l < LL; ++l) mu = fmaxf(mu, U[l]);
    float den = 0.f;
    for (int l = 0; l < LL; ++l) den = den + expf(U[l] - mu);
    float part = 0.f;
    for (int l = 0; l < LL; ++l) part += acl[l] * (expf(U[l] - mu) / den) * term[l];

    __shared__ float red[256];
    red[threadIdx.x] = part;
    __syncthreads();
    for (int s2 = 128; s2 > 0; s2 >>= 1) {
        if (threadIdx.x < s2) red[threadIdx.x] += red[threadIdx.x + s2];
        __syncthreads();
    }
    if (threadIdx.x == 0)
        __hip_atomic_fetch_add(out, red[0], __ATOMIC_RELAXED, __HIP_MEMORY_SCOPE_AGENT);
}

extern "C" void kernel_launch(void* const* d_in, const int* in_sizes, int n_in,
                              void* d_out, int out_size, void* d_ws, size_t ws_size,
                              hipStream_t stream) {
    (void)in_sizes; (void)n_in; (void)out_size; (void)ws_size;
    const float* Y      = (const float*)d_in[0];
    const float* access = (const float*)d_in[1];
    const float* beta   = (const float*)d_in[2];
    const float* alpha  = (const float*)d_in[3];
    const float* theta  = (const float*)d_in[4];
    const float* pi     = (const float*)d_in[5];
    const float* toff   = (const float*)d_in[6];
    const float* coff   = (const float*)d_in[7];
    float* ws = (float*)d_ws;

    hipMemsetAsync((char*)d_ws + OFF_E,    0, (size_t)KK * LL * CC * 4, stream);
    hipMemsetAsync((char*)d_ws + OFF_SUMY, 0, (size_t)KK * CC * 4,      stream);
    hipMemsetAsync(d_out, 0, sizeof(float), stream);

    prep_kernel<<<1, 64, 0, stream>>>(beta, alpha, theta, pi, ws);
    warp_kernel<<<dim3(CC, 16), 256, 0, stream>>>(toff, coff, ws);
    gemm_kernel<<<dim3(8, 50), 256, 0, stream>>>(Y, ws);
    finalize_kernel<<<32, 256, 0, stream>>>(access, ws, (float*)d_out);
}
